// SignatureGRU_81595788689886
// MI455X (gfx1250) — matrix operand order repaired and checked by reading, not compiled, not run on hardware
//
#include <hip/hip_runtime.h>
#include <hip/hip_bf16.h>
#include <math.h>

// SignatureGRU for MI455X (gfx1250): f16 WMMA GEMMs with fragment-packed
// operands (2x global_load_b128 per fragment), dual-accumulator waves for
// A-reuse, and a register-only fused GRU scan step (no LDS on critical path).

typedef _Float16 half8  __attribute__((ext_vector_type(8)));
typedef _Float16 half16 __attribute__((ext_vector_type(16)));
typedef float    float8 __attribute__((ext_vector_type(8)));

#define B_   64
#define T_   512
#define F_   256
#define U_   512
#define D_   5
#define SIG_ 30

__device__ __forceinline__ float sigmoidf_(float x) { return 1.0f / (1.0f + expf(-x)); }

// ---------------------------------------------------------------------------
// Fragment-packed operand layout (ISA 7.12.2, wave32, 16x16x32 f16):
// one fragment = 32 lanes x 16 f16 = 1KB; lane l owns the 32B at offset l*32.
// Loading a fragment = two coalesced global_load_b128 per lane.
// ---------------------------------------------------------------------------

__device__ __forceinline__ half16 load_frag(const _Float16* __restrict__ P, size_t tile) {
  int lane = threadIdx.x & 31;
  const half8* p = (const half8*)(P + (tile * 32 + lane) * 16);
  half8 lo = p[0];
  half8 hi = p[1];
  half16 r;
#pragma unroll
  for (int i = 0; i < 8; ++i) { r[i] = lo[i]; r[i + 8] = hi[i]; }
  return r;
}

// A-fragment packing rule: (m,k) -> tile (mt,kt); lane = half*16 + m%16 where
// half = (k>>3)&1; element i = (k&7) + ((k&16)?8:0).
__device__ __forceinline__ size_t a_pack_idx(int m, int k, int KT) {
  int mt = m >> 4, mm = m & 15;
  int kt = k >> 5, kk = k & 31;
  int hf = (kk >> 3) & 1;
  int i  = (kk & 7) + ((kk & 16) ? 8 : 0);
  int lane = hf * 16 + mm;
  return (((size_t)mt * KT + kt) * 32 + lane) * 16 + i;
}

// B-fragment packing rule: (k,n) -> tile (kt,nt); lane = half*16 + n%16 where
// half = (k>>3)&1; element i = (k&7) + ((k&16)?8:0).
__device__ __forceinline__ size_t b_pack_idx(int k, int n, int NT) {
  int kt = k >> 5, kk = k & 31;
  int hf = (kk >> 3) & 1;
  int i  = (kk & 7) + ((kk & 16) ? 8 : 0);
  int nt = n >> 4, nn = n & 15;
  int lane = hf * 16 + nn;
  return (((size_t)kt * NT + nt) * 32 + lane) * 16 + i;
}

// C/D store: 16x16 f32, VGPR r holds M = 8*half + r, N = lane%16.
__device__ __forceinline__ void store_c_frag(float* __restrict__ C, int ldc,
                                             int m0, int n0, float8 c) {
  int lane = threadIdx.x & 31;
  int n = n0 + (lane & 15);
  int h = lane >> 4;
#pragma unroll
  for (int r = 0; r < 8; ++r) C[(size_t)(m0 + h * 8 + r) * ldc + n] = c[r];
}

// ---------------------------------------------------------------------------
// Pack kernels: f32 row-major -> f16 fragment-packed
// ---------------------------------------------------------------------------

__global__ void pack_a_f16(const float* __restrict__ in, _Float16* __restrict__ out,
                           int M, int K) {
  size_t idx = (size_t)blockIdx.x * blockDim.x + threadIdx.x;
  if (idx >= (size_t)M * K) return;
  int m = (int)(idx / K), k = (int)(idx % K);
  out[a_pack_idx(m, k, K >> 5)] = (_Float16)in[idx];
}

__global__ void pack_b_f16(const float* __restrict__ in, _Float16* __restrict__ out,
                           int K, int N) {
  size_t idx = (size_t)blockIdx.x * blockDim.x + threadIdx.x;
  if (idx >= (size_t)K * N) return;
  int k = (int)(idx / N), n = (int)(idx % N);
  out[b_pack_idx(k, n, N >> 4)] = (_Float16)in[idx];
}

// ---------------------------------------------------------------------------
// Precompute kernels (small, f32 VALU for precision)
// ---------------------------------------------------------------------------

// path[bt][d] = sum_f x[bt][f] * sig_proj[f][d]
__global__ void path_kernel(const float* __restrict__ x, const float* __restrict__ sp,
                            float* __restrict__ path) {
  __shared__ float sps[F_ * D_];
  for (int i = threadIdx.x; i < F_ * D_; i += blockDim.x) sps[i] = sp[i];
  __syncthreads();
  int bt = blockIdx.x * blockDim.x + threadIdx.x;
  if (bt >= B_ * T_) return;
  float acc[D_] = {0.f, 0.f, 0.f, 0.f, 0.f};
  const float* xr = x + (size_t)bt * F_;
  for (int f = 0; f < F_; ++f) {
    float xv = xr[f];
#pragma unroll
    for (int d = 0; d < D_; ++d) acc[d] += xv * sps[f * D_ + d];
  }
#pragma unroll
  for (int d = 0; d < D_; ++d) path[(size_t)bt * D_ + d] = acc[d];
}

// Streaming depth-2 signature + 1/t normalization. One wave per batch row.
__global__ void sig_kernel(const float* __restrict__ path, float* __restrict__ ns) {
  int b = blockIdx.x;
  int lane = threadIdx.x;
  __shared__ float dxs[D_], s1p[D_];
  float s1 = 0.f, s2 = 0.f;
  int i = lane / D_, j = lane % D_;
  if (lane < SIG_) ns[(size_t)b * T_ * SIG_ + lane] = 0.f;  // t=0 row is zero
  for (int t = 1; t < T_; ++t) {
    if (lane < D_) {
      dxs[lane] = path[((size_t)b * T_ + t) * D_ + lane] -
                  path[((size_t)b * T_ + t - 1) * D_ + lane];
      s1p[lane] = s1;  // S1 before this increment (Chen term)
    }
    __syncthreads();
    float inv = 1.0f / (float)t;
    size_t o = ((size_t)b * T_ + t) * SIG_;
    if (lane < D_ * D_) {
      s2 += s1p[i] * dxs[j] + 0.5f * dxs[i] * dxs[j];
      ns[o + D_ + lane] = s2 * inv;
    }
    if (lane < D_) {
      s1 += dxs[lane];
      ns[o + lane] = s1 * inv;
    }
    __syncthreads();
  }
}

// r[bt][u] = sigmoid( norm_sig[bt] . reset_kernel[:,u] + b_r[u] )
__global__ void rgate_kernel(const float* __restrict__ ns, const float* __restrict__ rk,
                             const float* __restrict__ bias, float* __restrict__ rg) {
  size_t idx = (size_t)blockIdx.x * blockDim.x + threadIdx.x;
  if (idx >= (size_t)B_ * T_ * U_) return;
  int u = (int)(idx % U_);
  size_t bt = idx / U_;
  const float* s = ns + bt * SIG_;
  float acc = bias[U_ + u];
#pragma unroll
  for (int k = 0; k < SIG_; ++k) acc += s[k] * rk[(size_t)k * U_ + u];
  rg[idx] = sigmoidf_(acc);
}

// xz = x @ input_kernel : (32768 x 256) @ (256 x 1024), packed-f16 WMMA.
// Block = 8 waves; wave w computes TWO 16x16 tiles (n-tiles s and s+32)
// sharing one A-fragment load per k-step.
__global__ void xz_gemm_kernel(const _Float16* __restrict__ Xp,   // packed A (MT=2048,KT=8)
                               const _Float16* __restrict__ Wip,  // packed B (KT=8,NT=64)
                               float* __restrict__ xz) {
  const int KT = F_ >> 5;        // 8
  const int NT = (2 * U_) >> 4;  // 64
  int wave = threadIdx.x >> 5;
  int lane = threadIdx.x & 31;
  int mt  = blockIdx.x;
  int nt0 = blockIdx.y * 8 + wave;  // 0..31
  int nt1 = nt0 + 32;
  float8 acc0 = {};
  float8 acc1 = {};
#pragma unroll
  for (int kt = 0; kt < KT; ++kt) {
    half16 a  = load_frag(Xp,  (size_t)mt * KT + kt);
    half16 b0 = load_frag(Wip, (size_t)kt * NT + nt0);
    half16 b1 = load_frag(Wip, (size_t)kt * NT + nt1);
    if (kt + 1 < KT)  // CDNA5 global_prefetch of next B fragment
      __builtin_prefetch((const void*)(Wip + (((size_t)(kt + 1) * NT + nt0) * 32 + lane) * 16), 0, 1);
    acc0 = __builtin_amdgcn_wmma_f32_16x16x32_f16(false, a, false, b0, (short)0, acc0,
                                                  false, false);
    acc1 = __builtin_amdgcn_wmma_f32_16x16x32_f16(false, a, false, b1, (short)0, acc1,
                                                  false, false);
  }
  store_c_frag(xz, 2 * U_, mt * 16, nt0 * 16, acc0);
  store_c_frag(xz, 2 * U_, mt * 16, nt1 * 16, acc1);
}

__global__ void zero_h_kernel(float* __restrict__ h32, _Float16* __restrict__ h16) {
  int i = blockIdx.x * blockDim.x + threadIdx.x;
  if (i < B_ * U_) { h32[i] = 0.f; h16[i] = (_Float16)0.f; }
}

// ---------------------------------------------------------------------------
// Fused GRU step: one WAVE owns BOTH the z-tile and the n-tile for (mt,ut),
// sharing the A (h) fragment each k-step. Both accumulators have identical
// C layouts, so gating for each (b,u) is register-local: no LDS, no barrier.
// Grid = 4 (batch m-tiles) x 32 (u-tiles) = 128 single-wave blocks.
// ---------------------------------------------------------------------------
__global__ void gru_step_kernel(const _Float16* __restrict__ Hp,   // packed h (MT=4,KT=16)
                                const _Float16* __restrict__ Wrp,  // packed W_rec (KT=16,NT=64)
                                const float*    __restrict__ H32,  // h_t f32 (row-major)
                                const float*    __restrict__ xz,   // (B,T,2U) f32
                                const float*    __restrict__ rg,   // (B,T,U) sigmoid'ed
                                const float*    __restrict__ bias, // (3U)
                                float*          __restrict__ H32o,
                                _Float16*       __restrict__ Hpo,  // packed h_{t+1}
                                int t) {
  const int KT = U_ >> 5;        // 16
  const int NT = (2 * U_) >> 4;  // 64
  int lane = threadIdx.x & 31;
  int mt = blockIdx.x >> 5;      // 0..3
  int ut = blockIdx.x & 31;      // 0..31
  int ntz = ut;                  // z-half column tile
  int ntn = 32 + ut;             // n-half column tile

  float8 accz = {};
  float8 accn = {};
#pragma unroll
  for (int kt = 0; kt < KT; ++kt) {
    half16 a  = load_frag(Hp,  (size_t)mt * KT + kt);
    half16 bz = load_frag(Wrp, (size_t)kt * NT + ntz);
    half16 bn = load_frag(Wrp, (size_t)kt * NT + ntn);
    if (kt + 1 < KT)
      __builtin_prefetch((const void*)(Wrp + (((size_t)(kt + 1) * NT + ntz) * 32 + lane) * 16), 0, 1);
    accz = __builtin_amdgcn_wmma_f32_16x16x32_f16(false, a, false, bz, (short)0, accz,
                                                  false, false);
    accn = __builtin_amdgcn_wmma_f32_16x16x32_f16(false, a, false, bn, (short)0, accn,
                                                  false, false);
  }

  // Register-local gating: element r holds (b = mt*16 + 8*(lane>>4) + r,
  //                                         u = ut*16 + lane%16) in BOTH tiles.
  int u  = ut * 16 + (lane & 15);
  int b0 = mt * 16 + (lane >> 4) * 8;
  float bz_ = bias[u];
  float bn_ = bias[2 * U_ + u];
#pragma unroll
  for (int r = 0; r < 8; ++r) {
    int b = b0 + r;
    size_t bt = (size_t)b * T_ + t;
    float hz = accz[r];                 // h_rec[:, u]   (raw, reused in candidate)
    float hn = accn[r];                 // h_rec[:, U+u]
    float xzz = xz[bt * (2 * U_) + u];
    float xzn = xz[bt * (2 * U_) + U_ + u];
    float rv  = rg[bt * U_ + u];
    float z = sigmoidf_(hz + xzz + bz_);
    float n = tanhf(hn + xzn + bn_ + rv * hz);
    float hold = H32[(size_t)b * U_ + u];
    float hnew = z * hold + (1.0f - z) * n;
    H32o[(size_t)b * U_ + u] = hnew;
    Hpo[a_pack_idx(b, u, KT)] = (_Float16)hnew;  // next step's A-fragment layout
  }
}

// ---------------------------------------------------------------------------
extern "C" void kernel_launch(void* const* d_in, const int* in_sizes, int n_in,
                              void* d_out, int out_size, void* d_ws, size_t ws_size,
                              hipStream_t stream) {
  const float* x    = (const float*)d_in[0];  // (B,T,F)
  const float* sp   = (const float*)d_in[1];  // (F,D)
  const float* rk   = (const float*)d_in[2];  // (SIG,U)
  const float* wi   = (const float*)d_in[3];  // (F,2U)
  const float* wr   = (const float*)d_in[4];  // (U,2U)
  const float* bias = (const float*)d_in[5];  // (3U)

  char* p = (char*)d_ws;
  auto alloc = [&](size_t bytes) -> void* {
    void* q = (void*)p;
    p += (bytes + 255) & ~(size_t)255;
    return q;
  };
  _Float16* xp   = (_Float16*)alloc((size_t)B_ * T_ * F_ * 2);   // packed A for xz GEMM
  _Float16* wip  = (_Float16*)alloc((size_t)F_ * 2 * U_ * 2);    // packed B (input_kernel)
  _Float16* wrp  = (_Float16*)alloc((size_t)U_ * 2 * U_ * 2);    // packed B (recurrent)
  float*    path = (float*)alloc((size_t)B_ * T_ * D_ * 4);
  float*    ns   = (float*)alloc((size_t)B_ * T_ * SIG_ * 4);
  float*    rg   = (float*)alloc((size_t)B_ * T_ * U_ * 4);
  float*    xz   = (float*)alloc((size_t)B_ * T_ * 2 * U_ * 4);
  float*    h32a = (float*)alloc((size_t)B_ * U_ * 4);
  float*    h32b = (float*)alloc((size_t)B_ * U_ * 4);
  _Float16* hpa  = (_Float16*)alloc((size_t)B_ * U_ * 2);        // packed h (f16)
  _Float16* hpb  = (_Float16*)alloc((size_t)B_ * U_ * 2);
  (void)ws_size; (void)in_sizes; (void)n_in; (void)out_size;

  {
    size_t n = (size_t)B_ * T_ * F_;
    pack_a_f16<<<(unsigned)((n + 255) / 256), 256, 0, stream>>>(x, xp, B_ * T_, F_);
  }
  {
    size_t n = (size_t)F_ * 2 * U_;
    pack_b_f16<<<(unsigned)((n + 255) / 256), 256, 0, stream>>>(wi, wip, F_, 2 * U_);
  }
  {
    size_t n = (size_t)U_ * 2 * U_;
    pack_b_f16<<<(unsigned)((n + 255) / 256), 256, 0, stream>>>(wr, wrp, U_, 2 * U_);
  }

  path_kernel<<<(B_ * T_ + 255) / 256, 256, 0, stream>>>(x, sp, path);
  sig_kernel<<<B_, 32, 0, stream>>>(path, ns);
  {
    size_t tot = (size_t)B_ * T_ * U_;
    rgate_kernel<<<(unsigned)((tot + 255) / 256), 256, 0, stream>>>(ns, rk, bias, rg);
  }
  {
    dim3 g((B_ * T_) / 16, 4);  // 2048 x 4; 8 waves/block, 2 n-tiles/wave
    xz_gemm_kernel<<<g, 256, 0, stream>>>(xp, wip, xz);
  }
  zero_h_kernel<<<(B_ * U_ + 255) / 256, 256, 0, stream>>>(h32a, hpa);

  float*    hi32 = h32a; _Float16* hip = hpa;
  float*    ho32 = h32b; _Float16* hop = hpb;
  for (int t = 0; t < T_; ++t) {
    float* outp = (t == T_ - 1) ? (float*)d_out : ho32;  // final h -> d_out directly
    gru_step_kernel<<<128, 32, 0, stream>>>(hip, wrp, hi32, xz, rg, bias, outp, hop, t);
    float*    t32 = hi32; hi32 = ho32; ho32 = t32;
    _Float16* tp  = hip;  hip  = hop;  hop  = tp;
  }
}
